// SSCL_SAMAttention_40973988004756
// MI455X (gfx1250) — matrix operand order repaired and checked by reading, not compiled, not run on hardware
//
#include <hip/hip_runtime.h>

typedef __attribute__((ext_vector_type(16))) _Float16 v16h;
typedef __attribute__((ext_vector_type(8)))  _Float16 v8h;
typedef __attribute__((ext_vector_type(8)))  float    v8f;

#define B_   2
#define H_   8
#define NT   2048
#define D_   64
#define NQ   (B_ * H_ * NT * D_)      // 2,097,152 elements per (B,H,N,D) tensor
#define SPH  2056                     // f16 LDS strip pitch in halves (4112 B/row, ≡4 dwords mod 64)

// dynamic LDS: two f16 strips + small float arrays
#define LDS_BYTES (2 * 16 * SPH * 2 + 6 * 16 * 4)

// workspace layout (bytes): [0,72) stats+denom floats; f16 tensors at 128B alignment
#define WS_KH_OFF   128
#define WS_X2H_OFF  (WS_KH_OFF  + NQ * 2)
#define WS_VTH_OFF  (WS_X2H_OFF + NQ * 2)

// ---------------- WMMA helpers ----------------

__device__ __forceinline__ v8f wmma_f16(v16h a, v16h b, v8f c) {
    return __builtin_amdgcn_wmma_f32_16x16x32_f16(false, a, false, b, (short)0, c, false, false);
}

// A-matrix 16x32 f16 frag from row-major f32 (row stride 64), scaled (setup only).
__device__ __forceinline__ v16h load_a_f32(const float* base, int kk0, int lane, float scale) {
    const float* p = base + (size_t)(lane & 15) * D_ + kk0 + ((lane >> 4) << 4);
    v16h a;
#pragma unroll
    for (int i = 0; i < 8; ++i) a[i]     = (_Float16)(p[i]      * scale);
#pragma unroll
    for (int i = 0; i < 8; ++i) a[8 + i] = (_Float16)(p[8 + i] * scale);
    return a;
}

// ---------------- prep kernels ----------------

__global__ void init_ws_kernel(float* ws) {
    if (threadIdx.x < 18) ws[threadIdx.x] = 0.0f;
}

// f32 -> f16 copies of K and X2 (B operands of the score GEMMs)
__global__ __launch_bounds__(256) void convert_kernel(const float* __restrict__ k,
                                                      const float* __restrict__ x2,
                                                      _Float16* __restrict__ kh,
                                                      _Float16* __restrict__ x2h) {
    size_t i = (size_t)blockIdx.x * 256 + threadIdx.x;
    kh[i]  = (_Float16)k[i];
    x2h[i] = (_Float16)x2[i];
}

// V -> f16 transposed per (b,h): vth[bh][d][m] = v[bh][m][d]
__global__ __launch_bounds__(256) void vtrans_kernel(const float* __restrict__ v,
                                                     _Float16* __restrict__ vth) {
    int bd = blockIdx.x;              // bh*64 + d
    int bh = bd >> 6;
    int d  = bd & 63;
    const float* vb = v + (size_t)bh * NT * D_ + d;
    _Float16* o = vth + (size_t)bd * NT;
    for (int m = threadIdx.x; m < NT; m += 256)
        o[m] = (_Float16)vb[(size_t)m * D_];
}

// global sum / sumsq of dist[b,n,m]; one block per (b,n)
__global__ __launch_bounds__(256) void stats_kernel(const float* __restrict__ dq,
                                                    const float* __restrict__ dk,
                                                    float* __restrict__ ws) {
    int bn = blockIdx.x;
    int b  = bn >> 11;
    int n  = bn & (NT - 1);
    const float* p = dq + ((size_t)b * NT + n) * 4;
    float x0 = p[0], x1 = p[1], x2 = p[2];
    float xn = x0 * x0 + x1 * x1 + x2 * x2;
    const float* dkb = dk + (size_t)b * NT * 4;
    float s = 0.f, s2 = 0.f;
    for (int m = threadIdx.x; m < NT; m += 256) {
        const float* py = dkb + (size_t)m * 4;
        float y0 = py[0], y1 = py[1], y2 = py[2];
        float d = xn + y0 * y0 + y1 * y1 + y2 * y2 - 2.f * (x0 * y0 + x1 * y1 + x2 * y2);
        s += d; s2 += d * d;
    }
    __shared__ float sh[256], sh2[256];
    sh[threadIdx.x] = s; sh2[threadIdx.x] = s2;
    __syncthreads();
    for (int st = 128; st > 0; st >>= 1) {
        if ((int)threadIdx.x < st) { sh[threadIdx.x]  += sh[threadIdx.x + st];
                                     sh2[threadIdx.x] += sh2[threadIdx.x + st]; }
        __syncthreads();
    }
    if (threadIdx.x == 0) { atomicAdd(&ws[0], sh[0]); atomicAdd(&ws[1], sh2[0]); }
}

// denom_c[bh] = sum_n softplus(c[bh,n]); one block per bh
__global__ __launch_bounds__(256) void denom_kernel(const float* __restrict__ c,
                                                    float* __restrict__ ws) {
    int bh = blockIdx.x;
    const float* p = c + (size_t)bh * NT;
    float s = 0.f;
    for (int n = threadIdx.x; n < NT; n += 256) {
        float x = p[n];
        s += (x > 20.f) ? x : log1pf(expf(x));
    }
    __shared__ float sh[256];
    sh[threadIdx.x] = s;
    __syncthreads();
    for (int st = 128; st > 0; st >>= 1) {
        if ((int)threadIdx.x < st) sh[threadIdx.x] += sh[threadIdx.x + st];
        __syncthreads();
    }
    if (threadIdx.x == 0) ws[2 + bh] = s = sh[0];
}

// ---------------- main fused kernel ----------------
// grid: (B*H, N/16), block: 128 threads = 4 wave32
__global__ __launch_bounds__(128) void fused_attn_kernel(
        const float* __restrict__ q,   const float* __restrict__ x1,
        const _Float16* __restrict__ kh, const _Float16* __restrict__ x2h,
        const _Float16* __restrict__ vth,
        const float* __restrict__ dq,  const float* __restrict__ dk,
        const float* __restrict__ lin_w, const float* __restrict__ lin_b,
        const float* __restrict__ bn_g,  const float* __restrict__ bn_b,
        const float* __restrict__ alpha, const float* __restrict__ ws,
        float* __restrict__ out, float* __restrict__ scores_out) {
    extern __shared__ _Float16 ldsh[];
    _Float16* s1h = ldsh;                 // 16*SPH raw relu S1 (f16)
    _Float16* s2h = ldsh + 16 * SPH;      // 16*SPH raw relu S2 (f16)
    float* fext    = (float*)(ldsh + 2 * 16 * SPH);
    float* rowsum1 = fext;                // 16
    float* rowsum2 = fext + 16;
    float* prodsum = fext + 32;
    float* scale1  = fext + 48;
    float* scale2  = fext + 64;
    float* corr    = fext + 80;

    const int bh   = blockIdx.x;
    const int b    = bh >> 3;
    const int h    = bh & 7;
    const int n0   = blockIdx.y * 16;
    const int tid  = threadIdx.x;
    const int lane = tid & 31;
    const int wid  = tid >> 5;
    const int half = lane >> 4;
    const int col  = lane & 15;

    const float* qb  = q  + (size_t)bh * NT * D_;
    const float* x1b = x1 + (size_t)bh * NT * D_;
    const _Float16* khb  = kh  + (size_t)bh * NT * D_;
    const _Float16* x2hb = x2h + (size_t)bh * NT * D_;
    const float* dqb = dq + (size_t)b * NT * 4;
    const float* dkb = dk + (size_t)b * NT * 4;

    const float invBNM = 1.0f / ((float)B_ * NT * NT);
    const float mean  = ws[0] * invBNM;
    const float var   = ws[1] * invBNM - mean * mean;
    const float rstd  = rsqrtf(var + 1e-5f);
    const float gamma = bn_g[0], beta = bn_b[0];
    const float wgt   = lin_w[h], bia = lin_b[h];
    const float al    = alpha[h];
    const float denom = ws[2 + bh];

    // distmap affine folded to  A*dist + C  (per (b,h) scalars)
    const float A = rstd * gamma * wgt;
    const float C = (beta - mean * rstd * gamma) * wgt + bia;

    if (tid < 16) {
        rowsum1[tid] = 0.f; rowsum2[tid] = 0.f; prodsum[tid] = 0.f;
    }

    // register-resident row data for the distmap (rows half*8 + j)
    float qm0[8], qm1[8], qm2[8], qnA[8], q3r[8];
#pragma unroll
    for (int j = 0; j < 8; ++j) {
        const float* p = dqb + (size_t)(n0 + half * 8 + j) * 4;
        float a0 = p[0], a1 = p[1], a2 = p[2], a3 = p[3];
        qm0[j] = -2.f * a0; qm1[j] = -2.f * a1; qm2[j] = -2.f * a2;
        qnA[j] = A * (a0 * a0 + a1 * a1 + a2 * a2) + C;
        q3r[j] = a3;
    }

    // A fragments for this 16-row strip (1/sqrt(64) folded in)
    const v16h aq0 = load_a_f32(qb  + (size_t)n0 * D_, 0,  lane, 0.125f);
    const v16h aq1 = load_a_f32(qb  + (size_t)n0 * D_, 32, lane, 0.125f);
    const v16h ax0 = load_a_f32(x1b + (size_t)n0 * D_, 0,  lane, 0.125f);
    const v16h ax1 = load_a_f32(x1b + (size_t)n0 * D_, 32, lane, 0.125f);

    __syncthreads();

    float rp1[8], rp2[8];
#pragma unroll
    for (int j = 0; j < 8; ++j) { rp1[j] = 0.f; rp2[j] = 0.f; }

    // ---- Phase 1: score tiles across the full M range ----
    for (int mt = wid; mt < NT / 16; mt += 4) {
        const int m0 = mt * 16;
        const int m  = m0 + col;
        // B-fragments: contiguous 32B f16 loads, no conversion
        const _Float16* kp = khb  + (size_t)m * D_ + half * 16;
        const _Float16* xp = x2hb + (size_t)m * D_ + half * 16;
        v16h bk0 = *(const v16h*)(kp);
        v16h bk1 = *(const v16h*)(kp + 32);
        v16h bx0 = *(const v16h*)(xp);
        v16h bx1 = *(const v16h*)(xp + 32);

        v8f acc1 = {};
        acc1 = wmma_f16(aq0, bk0, acc1);
        acc1 = wmma_f16(aq1, bk1, acc1);
        v8f acc2 = {};
        acc2 = wmma_f16(ax0, bx0, acc2);
        acc2 = wmma_f16(ax1, bx1, acc2);

        const float* pk = dkb + (size_t)m * 4;
        float y0 = pk[0], y1 = pk[1], y2 = pk[2], y3 = pk[3];
        float e  = A * (y0 * y0 + y1 * y1 + y2 * y2);
        float ya = A * y0, yb = A * y1, yc = A * y2;

#pragma unroll
        for (int j = 0; j < 8; ++j) {
            const int r = half * 8 + j;
            float t = qnA[j] + e;                 // A*(qn+ynorm) + C
            t = fmaf(qm0[j], ya, t);
            t = fmaf(qm1[j], yb, t);
            t = fmaf(qm2[j], yc, t);
            float dm  = t * (q3r[j] * y3);
            float s1v = fmaxf(acc1[j] + dm, 0.f);
            float s2v = fmaxf(acc2[j] + dm, 0.f);
            s1h[(size_t)r * SPH + m] = (_Float16)s1v;
            s2h[(size_t)r * SPH + m] = (_Float16)s2v;
            rp1[j] += s1v; rp2[j] += s2v;
        }
    }
    // reduce per-row partials within 16-lane halves, one atomic per row per wave
#pragma unroll
    for (int j = 0; j < 8; ++j) {
        float v1 = rp1[j], v2 = rp2[j];
        for (int off = 1; off < 16; off <<= 1) {
            v1 += __shfl_xor(v1, off);
            v2 += __shfl_xor(v2, off);
        }
        if (col == 0) {
            atomicAdd(&rowsum1[half * 8 + j], v1);
            atomicAdd(&rowsum2[half * 8 + j], v2);
        }
    }
    __syncthreads();

    if (tid < 16) {
        scale1[tid] = 1.f / (rowsum1[tid] + denom + 1e-9f);
        scale2[tid] = 1.f / (rowsum2[tid] + denom + 1e-9f);
    }
    __syncthreads();

    // ---- Phase 3: write normalized scores once; reduce sum(S1raw*S2raw) ----
    float* srow_base = scores_out + ((size_t)bh * NT + n0) * NT;
    for (int r = 0; r < 16; ++r) {
        const float sc = scale1[r];
        float part = 0.f;
        const v8h* p1 = (const v8h*)(s1h + (size_t)r * SPH);
        const v8h* p2 = (const v8h*)(s2h + (size_t)r * SPH);
        float4* po = (float4*)(srow_base + (size_t)r * NT);
        for (int c8 = tid; c8 < NT / 8; c8 += 128) {
            v8h a = p1[c8];
            v8h bq = p2[c8];
            float fa[8], fb[8];
#pragma unroll
            for (int i = 0; i < 8; ++i) { fa[i] = (float)a[i]; fb[i] = (float)bq[i]; }
            float4 o0, o1;
            o0.x = fa[0] * sc; o0.y = fa[1] * sc; o0.z = fa[2] * sc; o0.w = fa[3] * sc;
            o1.x = fa[4] * sc; o1.y = fa[5] * sc; o1.z = fa[6] * sc; o1.w = fa[7] * sc;
            po[c8 * 2]     = o0;
            po[c8 * 2 + 1] = o1;
#pragma unroll
            for (int i = 0; i < 8; ++i) part += fa[i] * fb[i];
        }
        for (int off = 1; off < 32; off <<= 1) part += __shfl_xor(part, off);
        if (lane == 0) atomicAdd(&prodsum[r], part);
    }
    __syncthreads();
    if (tid < 16) corr[tid] = al * scale1[tid] * scale2[tid] * prodsum[tid];
    __syncthreads();

    // ---- Phase 4: out = scale1 * (S1raw @ V) + corr ----
    // wave wid owns D-columns [wid*16, wid*16+16)
    const int d0 = wid * 16;
    const _Float16* vtp  = vth + ((size_t)bh * D_ + d0 + col) * NT + half * 16;
    const _Float16* arow = s1h + (size_t)col * SPH + half * 16;   // A row = lane&15
    v8f acc = {};
    for (int kk = 0; kk < NT; kk += 32) {
        v8h lo = *(const v8h*)(arow + kk);
        v8h hi = *(const v8h*)(arow + kk + 8);
        v16h af;
#pragma unroll
        for (int i = 0; i < 8; ++i) { af[i] = lo[i]; af[8 + i] = hi[i]; }
        v16h bf = *(const v16h*)(vtp + kk);
        acc = wmma_f16(af, bf, acc);
    }
    float* ob = out + ((size_t)bh * NT + n0) * D_;
#pragma unroll
    for (int j = 0; j < 8; ++j) {
        const int r = half * 8 + j;
        ob[(size_t)r * D_ + d0 + col] = acc[j] * scale1[r] + corr[r];
    }
}

// ---------------- launcher ----------------

extern "C" void kernel_launch(void* const* d_in, const int* in_sizes, int n_in,
                              void* d_out, int out_size, void* d_ws, size_t ws_size,
                              hipStream_t stream) {
    const float* q     = (const float*)d_in[0];
    const float* k     = (const float*)d_in[1];
    const float* v     = (const float*)d_in[2];
    const float* x1    = (const float*)d_in[3];
    const float* x2    = (const float*)d_in[4];
    const float* c     = (const float*)d_in[5];
    const float* dq    = (const float*)d_in[6];
    const float* dk    = (const float*)d_in[7];
    const float* lw    = (const float*)d_in[8];
    const float* lb    = (const float*)d_in[9];
    const float* bng   = (const float*)d_in[10];
    const float* bnb   = (const float*)d_in[11];
    const float* alpha = (const float*)d_in[12];

    float*    ws  = (float*)d_ws;
    _Float16* kh  = (_Float16*)((char*)d_ws + WS_KH_OFF);
    _Float16* x2h = (_Float16*)((char*)d_ws + WS_X2H_OFF);
    _Float16* vth = (_Float16*)((char*)d_ws + WS_VTH_OFF);

    float* out_ptr    = (float*)d_out;                        // (B,H,N,64)
    float* scores_ptr = out_ptr + (size_t)B_ * H_ * NT * D_;  // (B,H,N,M)

    init_ws_kernel<<<1, 32, 0, stream>>>(ws);
    stats_kernel<<<B_ * NT, 256, 0, stream>>>(dq, dk, ws);
    denom_kernel<<<B_ * H_, 256, 0, stream>>>(c, ws);
    convert_kernel<<<NQ / 256, 256, 0, stream>>>(k, x2, kh, x2h);
    vtrans_kernel<<<B_ * H_ * D_, 256, 0, stream>>>(v, vth);

    dim3 grid(B_ * H_, NT / 16);
    fused_attn_kernel<<<grid, 128, LDS_BYTES, stream>>>(
        q, x1, kh, x2h, vth, dq, dk, lw, lb, bng, bnb, alpha, ws,
        out_ptr, scores_ptr);
}
